// ConvAttention_56246891708756
// MI455X (gfx1250) — matrix operand order repaired and checked by reading, not compiled, not run on hardware
//
#include <hip/hip_runtime.h>
#include <hip/hip_bf16.h>

typedef _Float16 f16;
typedef __attribute__((ext_vector_type(16))) _Float16 v16h;
typedef __attribute__((ext_vector_type(8)))  _Float16 v8h;
typedef __attribute__((ext_vector_type(8)))  float    v8f;

#define Hh 8
#define Dd 64
#define Bb 8
#define Nn 1024
#define HD 512
#define EPS 1e-5f

// LDS prob tile geometry: row stride 40 halves (80B = 5*16B -> b128-aligned
// rows; 20-dword stride makes 16 lanes' 4-dword reads hit all 64 banks once)
#define PROW 40
#define PHEAD (16 * PROW)

__device__ __forceinline__ v16h load_v16(const f16* lo, const f16* hi) {
  union { v16h v; v8h h[2]; } u;
  u.h[0] = *(const v8h*)lo;
  u.h[1] = *(const v8h*)hi;
  return u.v;
}

__device__ __forceinline__ v8f zero_v8f() {
  v8f z;
#pragma unroll
  for (int i = 0; i < 8; ++i) z[i] = 0.f;
  return z;
}

__device__ __forceinline__ v8h zero_v8h() {
  v8h z;
#pragma unroll
  for (int i = 0; i < 8; ++i) z[i] = (f16)0.f;
  return z;
}

__device__ __forceinline__ v8f wmma_f16(v16h a, v16h b, v8f c) {
  return __builtin_amdgcn_wmma_f32_16x16x32_f16(false, a, false, b,
                                                (short)0, c, false, false);
}

// K-tile fragments: B-layout for QK^T, two j-subtiles x two K(d) chunks
struct KFrag { v16h f[2][2]; };
__device__ __forceinline__ KFrag loadK(const f16* kh, int j0, int lm, int lg) {
  KFrag kb;
#pragma unroll
  for (int st = 0; st < 2; ++st)
#pragma unroll
    for (int kc = 0; kc < 2; ++kc) {
      const f16* p = kh + (size_t)(j0 + st * 16 + lm) * Dd + kc * 32 + lg * 8;
      kb.f[st][kc] = load_v16(p, p + 16);
    }
  return kb;
}

// V-tile fragments: B-layout for P'@V from transposed v
struct VFrag { v16h f[4]; };
__device__ __forceinline__ VFrag loadV(const f16* vh, int j0, int lm, int lg) {
  VFrag vb;
#pragma unroll
  for (int nt = 0; nt < 4; ++nt) {
    const f16* p = vh + (size_t)(nt * 16 + lm) * Nn + j0 + lg * 8;
    vb.f[nt] = load_v16(p, p + 16);
  }
  return vb;
}

// ---------------------------------------------------------------------------
// Kernel 1: depthwise 3x3 + BN + pointwise head mix -> q, k row major f16,
// v transposed [b][h][d][n] f16 so attn@V B-fragments load contiguously.
// ---------------------------------------------------------------------------
__global__ __launch_bounds__(256)
void proj_kernel(const float* __restrict__ x,
                 const float* __restrict__ dw_w, const float* __restrict__ dw_b,
                 const float* __restrict__ bn_g, const float* __restrict__ bn_b,
                 const float* __restrict__ bn_m, const float* __restrict__ bn_v,
                 const float* __restrict__ pw_w, const float* __restrict__ pw_b,
                 f16* __restrict__ q, f16* __restrict__ k, f16* __restrict__ vT)
{
  int tid = threadIdx.x;
  int d  = tid & 63;
  int nn = tid >> 6;                  // 0..3
  int blk = blockIdx.x;               // Bb * (Nn/4)
  int b = blk >> 8;
  int n = ((blk & 255) << 2) + nn;

  float xin[3][Hh][3];
#pragma unroll
  for (int rn = 0; rn < 3; ++rn) {
    int nsrc = n + rn - 1;
    bool nok = (nsrc >= 0) && (nsrc < Nn);
    const float* xrow = x + ((size_t)b * Nn + (nok ? nsrc : 0)) * HD;
#pragma unroll
    for (int h = 0; h < Hh; ++h)
#pragma unroll
      for (int rd = 0; rd < 3; ++rd) {
        int dsrc = d + rd - 1;
        bool ok = nok && (dsrc >= 0) && (dsrc < Dd);
        xin[rn][h][rd] = ok ? xrow[h * Dd + dsrc] : 0.f;
      }
  }

#pragma unroll
  for (int i = 0; i < 3; ++i) {
    float u[Hh];
#pragma unroll
    for (int h = 0; h < Hh; ++h) {
      const float* w = dw_w + (i * Hh + h) * 9;
      float acc = 0.f;
#pragma unroll
      for (int rn = 0; rn < 3; ++rn)
#pragma unroll
        for (int rd = 0; rd < 3; ++rd)
          acc = fmaf(xin[rn][h][rd], w[rn * 3 + rd], acc);
      float inv = bn_g[i * Hh + h] * rsqrtf(bn_v[i * Hh + h] + EPS);
      u[h] = acc * inv + (dw_b[i * Hh + h] - bn_m[i * Hh + h]) * inv + bn_b[i * Hh + h];
    }
#pragma unroll
    for (int o = 0; o < Hh; ++o) {
      const float* pw = pw_w + (i * Hh + o) * Hh;
      float z = pw_b[i * Hh + o];
#pragma unroll
      for (int c = 0; c < Hh; ++c) z = fmaf(pw[c], u[c], z);
      f16 zh = (f16)z;
      size_t idx = (((size_t)b * Hh + o) * Nn + n) * Dd + d;
      if (i == 0)      q[idx] = zh;
      else if (i == 1) k[idx] = zh;
      else             vT[(((size_t)b * Hh + o) * Dd + d) * Nn + n] = zh;
    }
  }
}

// ---------------------------------------------------------------------------
// Kernel 2: per-(b,h,d) column sums of v (re-attention bias rank-1 term)
// ---------------------------------------------------------------------------
__global__ __launch_bounds__(256)
void vsum_kernel(const f16* __restrict__ vT, float* __restrict__ vsum) {
  int wave = (int)((blockIdx.x * 256 + threadIdx.x) >> 5);   // 0..4095 rows
  int lane = threadIdx.x & 31;
  const v8h* row = (const v8h*)(vT + (size_t)wave * Nn) + lane;  // 128 v8h/row
  float s = 0.f;
#pragma unroll
  for (int t = 0; t < 4; ++t) {
    v8h v = row[t * 32];
#pragma unroll
    for (int e = 0; e < 8; ++e) s += (float)v[e];
  }
#pragma unroll
  for (int m = 16; m; m >>= 1) s += __shfl_xor(s, m, 32);
  if (lane == 0) vsum[wave] = s;
}

// ---------------------------------------------------------------------------
// Kernel 3: f32 -> f16 weight conversion for the output GEMM
// ---------------------------------------------------------------------------
__global__ __launch_bounds__(256)
void wcvt_kernel(const float* __restrict__ w, f16* __restrict__ wh, int n) {
  int i = blockIdx.x * 1024 + threadIdx.x;
#pragma unroll
  for (int t = 0; t < 4; ++t) {
    int j = i + t * 256;
    if (j < n) wh[j] = (f16)w[j];
  }
}

// ---------------------------------------------------------------------------
// Kernel 4: fused attention with cross-head re-attention.
// Block = (b, 16-row query tile); wave w = head w.
//   pass 1: WMMA QK^T, k-fragments double-buffered across j-tiles;
//           per-lane online softmax; one 16-lane butterfly at the end.
//   pass 2: stage-rotated loop: dots(t) -> probs->LDS -> issue k(t+1)+v(t)
//           loads -> packed-f16 8x8 head mix (hides loads) -> PV WMMA(t).
// Re-attention BN folded: W'[o,c]=s_o*re_w[o,c]; bias adds beta_o*colsum(v_o).
// ---------------------------------------------------------------------------
__global__ __launch_bounds__(256)
void attn_kernel(const f16* __restrict__ q, const f16* __restrict__ k,
                 const f16* __restrict__ vT, const float* __restrict__ vsum,
                 const float* __restrict__ re_w, const float* __restrict__ re_b,
                 const float* __restrict__ vn_g, const float* __restrict__ vn_b,
                 const float* __restrict__ vn_m, const float* __restrict__ vn_v,
                 f16* __restrict__ Oh)
{
  __shared__ _Float16 lds_p[Hh * PHEAD];   // 8 heads x 16 rows x 40 halves

  const int lane = threadIdx.x & 31;
  const int wv   = threadIdx.x >> 5;         // head
  const int lg   = lane >> 4;                // half-wave
  const int lm   = lane & 15;
  const int b    = blockIdx.x >> 6;
  const int i0   = (blockIdx.x & 63) << 4;

  // folded re-attention weights (v8h splats for packed mix)
  float so = vn_g[wv] * rsqrtf(vn_v[wv] + EPS);
  v8h wmix8[Hh];
#pragma unroll
  for (int c = 0; c < Hh; ++c) {
    f16 wc = (f16)(so * re_w[wv * Hh + c]);
#pragma unroll
    for (int e = 0; e < 8; ++e) wmix8[c][e] = wc;
  }
  float beta = so * (re_b[wv] - vn_m[wv]) + vn_b[wv];

  const f16* qh = q  + ((size_t)b * Hh + wv) * Nn * Dd;
  const f16* kh = k  + ((size_t)b * Hh + wv) * Nn * Dd;
  const f16* vh = vT + ((size_t)b * Hh + wv) * Dd * Nn;

  // resident Q fragments (K=64 -> two K=32 chunks)
  v16h afrag[2];
  {
    int row = i0 + lm;
#pragma unroll
    for (int kc = 0; kc < 2; ++kc) {
      const f16* p0 = qh + (size_t)row * Dd + kc * 32 + lg * 8;
      afrag[kc] = load_v16(p0, p0 + 16);
    }
  }

  const float scale = 0.35355339059327373f;  // H^-0.5
  float mrow[8], lrow[8];
#pragma unroll
  for (int r = 0; r < 8; ++r) { mrow[r] = -1e30f; lrow[r] = 0.f; }

  // ---------------- pass 1: per-lane online softmax stats ----------------
  {
    KFrag kb[2];
    kb[0] = loadK(kh, 0, lm, lg);
#pragma unroll 2
    for (int jt = 0; jt < 32; ++jt) {
      const int cur = jt & 1;
      // next tile's loads in flight during this tile's WMMAs (wraps to 0 at end)
      kb[cur ^ 1] = loadK(kh, ((jt + 1) & 31) << 5, lm, lg);
      v8f d0 = zero_v8f(), d1 = zero_v8f();
      d0 = wmma_f16(afrag[0], kb[cur].f[0][0], d0);
      d1 = wmma_f16(afrag[0], kb[cur].f[1][0], d1);
      d0 = wmma_f16(afrag[1], kb[cur].f[0][1], d0);
      d1 = wmma_f16(afrag[1], kb[cur].f[1][1], d1);
#pragma unroll
      for (int r = 0; r < 8; ++r) {
        float a0 = d0[r] * scale, a1 = d1[r] * scale;
        float mnew = fmaxf(mrow[r], fmaxf(a0, a1));
        lrow[r] = lrow[r] * __expf(mrow[r] - mnew)
                + __expf(a0 - mnew) + __expf(a1 - mnew);
        mrow[r] = mnew;
      }
    }
  }
  // single cross-lane combine (16-lane butterfly, outside the loop)
  float rl[8];
#pragma unroll
  for (int r = 0; r < 8; ++r) {
    float m = mrow[r], s = lrow[r];
#pragma unroll
    for (int msk = 1; msk < 16; msk <<= 1) {
      float om = __shfl_xor(m, msk, 32);
      float os = __shfl_xor(s, msk, 32);
      float mn = fmaxf(m, om);
      s = s * __expf(m - mn) + os * __expf(om - mn);
      m = mn;
    }
    mrow[r] = m;
    rl[r] = 1.f / s;
  }

  // ---------------- pass 2: dots -> probs -> mix -> P'@V ----------------
  v8f acc[4];
#pragma unroll
  for (int nt = 0; nt < 4; ++nt) acc[nt] = zero_v8f();

  {
    KFrag kb[2];
    kb[0] = loadK(kh, 0, lm, lg);
#pragma unroll 2
    for (int jt = 0; jt < 32; ++jt) {
      const int cur = jt & 1;
      const int j0 = jt << 5;

      v8f d0 = zero_v8f(), d1 = zero_v8f();
      d0 = wmma_f16(afrag[0], kb[cur].f[0][0], d0);
      d1 = wmma_f16(afrag[0], kb[cur].f[1][0], d1);
      d0 = wmma_f16(afrag[1], kb[cur].f[0][1], d0);
      d1 = wmma_f16(afrag[1], kb[cur].f[1][1], d1);

      // own head's normalized probs -> LDS (row r+8*lg, col lm / lm+16)
#pragma unroll
      for (int r = 0; r < 8; ++r) {
        int rr = r + 8 * lg;
        float p0 = __expf(d0[r] * scale - mrow[r]) * rl[r];
        float p1 = __expf(d1[r] * scale - mrow[r]) * rl[r];
        lds_p[wv * PHEAD + rr * PROW + lm]      = (f16)p0;
        lds_p[wv * PHEAD + rr * PROW + 16 + lm] = (f16)p1;
      }
      __syncthreads();

      // issue next k-tile and this tile's v-tile loads; the LDS mix below
      // (16x ds_load_b128 + 64x v_pk_fma) hides their latency
      kb[cur ^ 1] = loadK(kh, ((jt + 1) & 31) << 5, lm, lg);
      VFrag vb = loadV(vh, j0, lm, lg);

      // cross-head mix: per head 2x ds_load_b128 (conflict-free), packed fma
      v8h alo = zero_v8h(), ahi = zero_v8h();
#pragma unroll
      for (int c = 0; c < Hh; ++c) {
        v8h plo = *(const v8h*)&lds_p[c * PHEAD + lm * PROW + lg * 8];
        v8h phi = *(const v8h*)&lds_p[c * PHEAD + lm * PROW + 16 + lg * 8];
        alo += wmix8[c] * plo;
        ahi += wmix8[c] * phi;
      }
      __syncthreads();

      union { v16h v; v8h h[2]; } ap;
      ap.h[0] = alo;
      ap.h[1] = ahi;
#pragma unroll
      for (int nt = 0; nt < 4; ++nt)
        acc[nt] = wmma_f16(ap.v, vb.f[nt], acc[nt]);
    }
  }

  // epilogue: + beta * colsum(v), store f16 rows of [B*N, 512]
  f16* orow = Oh + ((size_t)b * Nn + i0) * HD + wv * Dd;
#pragma unroll
  for (int nt = 0; nt < 4; ++nt) {
    float vs = vsum[((size_t)b * Hh + wv) * Dd + nt * 16 + lm];
#pragma unroll
    for (int r = 0; r < 8; ++r) {
      float val = acc[nt][r] + beta * vs;
      orow[(size_t)(r + 8 * lg) * HD + nt * 16 + lm] = (f16)val;
    }
  }
}

// ---------------------------------------------------------------------------
// Kernel 5: output linear [8192,512] @ [512,512]^T + bias, f16 WMMA, f32 out,
// A/B fragments double-buffered across the 16 K-chunks.
// ---------------------------------------------------------------------------
__global__ __launch_bounds__(256)
void outgemm_kernel(const f16* __restrict__ A, const f16* __restrict__ Wh,
                    const float* __restrict__ bias, float* __restrict__ out)
{
  const int lane = threadIdx.x & 31;
  const int wv   = threadIdx.x >> 5;
  const int lg   = lane >> 4;
  const int lm   = lane & 15;
  const int rt   = blockIdx.x;       // 512 row tiles of 16
  const int row  = rt * 16 + lm;
  const int o0   = wv * 64;

  v8f acc[4];
#pragma unroll
  for (int nt = 0; nt < 4; ++nt) acc[nt] = zero_v8f();

  v16h af[2];
  v16h bf[2][4];
  {
    const f16* pa = A + (size_t)row * HD + lg * 8;
    af[0] = load_v16(pa, pa + 16);
#pragma unroll
    for (int nt = 0; nt < 4; ++nt) {
      const f16* pb = Wh + (size_t)(o0 + nt * 16 + lm) * HD + lg * 8;
      bf[0][nt] = load_v16(pb, pb + 16);
    }
  }
#pragma unroll 2
  for (int kc = 0; kc < 16; ++kc) {
    const int cur = kc & 1;
    const int kn = ((kc + 1) & 15) * 32;
    const f16* pa = A + (size_t)row * HD + kn + lg * 8;
    af[cur ^ 1] = load_v16(pa, pa + 16);
#pragma unroll
    for (int nt = 0; nt < 4; ++nt) {
      const f16* pb = Wh + (size_t)(o0 + nt * 16 + lm) * HD + kn + lg * 8;
      bf[cur ^ 1][nt] = load_v16(pb, pb + 16);
    }
#pragma unroll
    for (int nt = 0; nt < 4; ++nt)
      acc[nt] = wmma_f16(af[cur], bf[cur][nt], acc[nt]);
  }
#pragma unroll
  for (int nt = 0; nt < 4; ++nt) {
    float bz = bias[o0 + nt * 16 + lm];
#pragma unroll
    for (int r = 0; r < 8; ++r)
      out[(size_t)(rt * 16 + r + 8 * lg) * HD + o0 + nt * 16 + lm] = acc[nt][r] + bz;
  }
}

// ---------------------------------------------------------------------------
extern "C" void kernel_launch(void* const* d_in, const int* in_sizes, int n_in,
                              void* d_out, int out_size, void* d_ws, size_t ws_size,
                              hipStream_t stream)
{
  const float* x    = (const float*)d_in[0];
  const float* dw_w = (const float*)d_in[1];
  const float* dw_b = (const float*)d_in[2];
  const float* bn_g = (const float*)d_in[3];
  const float* bn_b = (const float*)d_in[4];
  const float* bn_m = (const float*)d_in[5];
  const float* bn_v = (const float*)d_in[6];
  const float* pw_w = (const float*)d_in[7];
  const float* pw_b = (const float*)d_in[8];
  const float* re_w = (const float*)d_in[9];
  const float* re_b = (const float*)d_in[10];
  const float* vn_g = (const float*)d_in[11];
  const float* vn_b = (const float*)d_in[12];
  const float* vn_m = (const float*)d_in[13];
  const float* vn_v = (const float*)d_in[14];
  const float* outw = (const float*)d_in[15];
  const float* outb = (const float*)d_in[16];

  char* ws = (char*)d_ws;
  const size_t szqkv = (size_t)Bb * Hh * Nn * Dd * sizeof(f16);   // 8.39 MB each
  f16*   q   = (f16*)(ws);
  f16*   kk  = (f16*)(ws + szqkv);
  f16*   vT  = (f16*)(ws + 2 * szqkv);
  f16*   Oh  = (f16*)(ws + 3 * szqkv);
  f16*   Wh  = (f16*)(ws + 3 * szqkv + (size_t)Bb * Nn * HD * sizeof(f16));
  float* vsp = (float*)(ws + 3 * szqkv + (size_t)Bb * Nn * HD * sizeof(f16)
                           + (size_t)HD * HD * sizeof(f16));

  proj_kernel<<<2048, 256, 0, stream>>>(x, dw_w, dw_b, bn_g, bn_b, bn_m, bn_v,
                                        pw_w, pw_b, q, kk, vT);
  vsum_kernel<<<512, 256, 0, stream>>>(vT, vsp);
  wcvt_kernel<<<256, 256, 0, stream>>>(outw, Wh, HD * HD);
  attn_kernel<<<512, 256, 0, stream>>>(q, kk, vT, vsp, re_w, re_b,
                                       vn_g, vn_b, vn_m, vn_v, Oh);
  outgemm_kernel<<<512, 256, 0, stream>>>(Oh, Wh, outb, (float*)d_out);
}
